// LSTM_74706661146781
// MI455X (gfx1250) — matrix operand order repaired and checked by reading, not compiled, not run on hardware
//
#include <hip/hip_runtime.h>

// ---------------------------------------------------------------------------
// Persistent-RNN LSTM for gfx1250 (MI455X), wave32 + WMMA bf16 + async LDS.
//   B=16, T=4096, I=H=512.  gates = x_t@Wih^T + h@Whh^T + b_ih + b_hh
// 32 blocks x 128 threads (4 waves). Block b owns hidden cols [16b,16b+16).
// Wave w computes gate w (i/f/g/o): one 16x16 C tile, K=512 twice
// (x-proj from bf16 copy of x, h-proj register-resident Whh panel).
// x_{t+1} tile and h_{t-1} are fetched with GLOBAL_LOAD_ASYNC_TO_LDS_B128
// (ASYNCcnt) overlapping the current step's WMMA work; hidden state crosses
// blocks via h_glb + a monotonic arrive/release grid barrier.
// ---------------------------------------------------------------------------

typedef __attribute__((ext_vector_type(16))) __bf16 v16bf;
typedef __attribute__((ext_vector_type(8)))  __bf16 v8bf;
typedef __attribute__((ext_vector_type(4)))  __bf16 v4bf;
typedef __attribute__((ext_vector_type(8)))  float  v8f;
typedef __attribute__((ext_vector_type(4)))  float  v4f;

#define B_   16
#define T_   4096
#define I_   512
#define H_   512
#define NBLK 32
#define NTHR 128   // 4 wave32

// Convert 16 contiguous fp32 -> v16bf (one B fragment K-run for this lane).
static __device__ __forceinline__ v16bf cvt16(const float* __restrict__ p) {
  v4f a = *(const v4f*)(p + 0);
  v4f b = *(const v4f*)(p + 4);
  v4f c = *(const v4f*)(p + 8);
  v4f d = *(const v4f*)(p + 12);
  v16bf r;
  r[0]=(__bf16)a.x;  r[1]=(__bf16)a.y;  r[2]=(__bf16)a.z;  r[3]=(__bf16)a.w;
  r[4]=(__bf16)b.x;  r[5]=(__bf16)b.y;  r[6]=(__bf16)b.z;  r[7]=(__bf16)b.w;
  r[8]=(__bf16)c.x;  r[9]=(__bf16)c.y;  r[10]=(__bf16)c.z; r[11]=(__bf16)c.w;
  r[12]=(__bf16)d.x; r[13]=(__bf16)d.y; r[14]=(__bf16)d.z; r[15]=(__bf16)d.w;
  return r;
}

// A fragment: per lane one contiguous 16-elem run at K = k0..k0+15
// (lanes 16-31 use k0 += 16). 16B ds loads.
static __device__ __forceinline__ v16bf ldsA(const __bf16* p) {
  v8bf lo = *(const v8bf*)(p);
  v8bf hi = *(const v8bf*)(p + 8);
  return __builtin_shufflevector(lo, hi,
      0,1,2,3,4,5,6,7,8,9,10,11,12,13,14,15);
}

static __device__ __forceinline__ float sigm(float v) {
  return 1.0f / (1.0f + __expf(-v));
}

// Async 16B copy: global (saddr + 32-bit voffset) -> LDS (32-bit lds offset).
// Tracked with ASYNCcnt; drained with s_wait_asynccnt.
static __device__ __forceinline__ void async_cp16(unsigned lds_off,
                                                  unsigned byte_off,
                                                  const void* base) {
  asm volatile("global_load_async_to_lds_b128 %0, %1, %2 offset:0"
               :: "v"(lds_off), "v"(byte_off), "s"(base) : "memory");
}
static __device__ __forceinline__ void wait_async0() {
  asm volatile("s_wait_asynccnt 0x0" ::: "memory");
}
static __device__ __forceinline__ unsigned lds_off_of(const void* p) {
  return (unsigned)(size_t)p;   // generic LDS addr: low 32 bits = LDS offset
}

__global__ void init_ctr_kernel(unsigned* c) { c[0] = 0u; c[1] = 0u; }

// fp32 -> bf16 bulk convert (used for W_ih and the whole x tensor; both then
// live in L2 / HBM as bf16 and are consumed by WMMA without per-step cvt).
__global__ void cvt_bf16_kernel(const float* __restrict__ w,
                                __bf16* __restrict__ o, int n) {
  int i = (blockIdx.x * blockDim.x + threadIdx.x) * 4;
  if (i < n) {
    v4f v = *(const v4f*)(w + i);
    v4bf b; b[0]=(__bf16)v.x; b[1]=(__bf16)v.y; b[2]=(__bf16)v.z; b[3]=(__bf16)v.w;
    *(v4bf*)(o + i) = b;
  }
}

__global__ __launch_bounds__(NTHR, 1)
void lstm_scan_kernel(const __bf16* __restrict__ x_bf,  // [B,T,I] bf16
                      const float* __restrict__ w_hh,
                      const float* __restrict__ b_ih,
                      const float* __restrict__ b_hh,
                      const __bf16* __restrict__ wih_bf,
                      float* __restrict__ out,
                      __bf16* __restrict__ h_glb,   // [16][512] bf16
                      unsigned* __restrict__ ctr)   // [0]=arrive [1]=release
{
  __shared__ __bf16 xA[2][B_ * I_];    // double-buffered x_t tile (bf16)
  __shared__ __bf16 hA[B_ * H_];       // h_{t-1}, bf16
  __shared__ float  gbuf[4][B_][16];   // gate preacts for this block
  __shared__ float  cst[B_][16];       // c state (block-local)
  __shared__ float  hl [B_][16];       // last h fp32 (for h_n)

  const int blk  = blockIdx.x;
  const int tid  = threadIdx.x;
  const int lane = tid & 31;
  const int gate = tid >> 5;           // wave id = gate index 0..3
  const int l15  = lane & 15;
  const int lhi  = lane >> 4;          // 0: low K half, 1: high K half

  for (int i = tid; i < B_ * H_; i += NTHR) hA[i] = (__bf16)0.0f;
  for (int i = tid; i < B_ * 16; i += NTHR) ((float*)cst)[i] = 0.0f;

  const int hcol = blk * 16 + l15;     // hidden column 0..511
  const int gcol = gate * H_ + hcol;   // gate column 0..2047
  const float bsum = b_ih[gcol] + b_hh[gcol];

  // Whh B-panel resident in registers: 16 fragments (K=512), bf16.
  v16bf whh[16];
  {
    const float* wr = w_hh + (size_t)gcol * H_;
#pragma unroll
    for (int kk = 0; kk < 16; ++kk)
      whh[kk] = cvt16(wr + kk * 32 + lhi * 16);
  }
  const __bf16* wir = wih_bf + (size_t)gcol * I_;  // loop-invariant: hoisted
                                                   // into registers by LLVM
  const unsigned xoff[2] = { lds_off_of(&xA[0][0]), lds_off_of(&xA[1][0]) };
  const unsigned hoff    = lds_off_of(&hA[0]);

  // Prologue: kick off async fetch of x_0 into buffer 0.
  for (int i = tid * 8; i < B_ * I_; i += NTHR * 8) {
    const int m = i >> 9, k = i & (I_ - 1);
    async_cp16(xoff[0] + (unsigned)i * 2u,
               (unsigned)(((m * T_ + 0) * I_ + k) * 2), x_bf);
  }

  unsigned* arrive  = ctr;
  unsigned* release = ctr + 1;
  __syncthreads();

  for (int t = 0; t < T_; ++t) {
    // ---- drain async copies (x_t tile, h_{t-1}); all waves then agree ----
    wait_async0();
    __syncthreads();

    // ---- prefetch x_{t+1} into the other buffer (overlaps all work) ----
    if (t + 1 < T_) {
      const unsigned dst = xoff[(t + 1) & 1];
      for (int i = tid * 8; i < B_ * I_; i += NTHR * 8) {
        const int m = i >> 9, k = i & (I_ - 1);
        async_cp16(dst + (unsigned)i * 2u,
                   (unsigned)(((m * T_ + (t + 1)) * I_ + k) * 2), x_bf);
      }
    }

    // ---- gates[16x16] = x_t@Wih^T + h@Whh^T : 32 WMMAs, pipelined A ----
    const __bf16* xp = &xA[t & 1][0];
    v8f acc = {0.f, 0.f, 0.f, 0.f, 0.f, 0.f, 0.f, 0.f};
    v16bf ax = ldsA(xp + l15 * I_ + lhi * 16);
    v16bf ah = ldsA(hA + l15 * H_ + lhi * 16);
#pragma unroll
    for (int kk = 0; kk < 16; ++kk) {
      v16bf axn = ax, ahn = ah;
      if (kk < 15) {                     // issue k+1 loads before k's WMMAs
        const int k1 = (kk + 1) * 32 + lhi * 16;
        axn = ldsA(xp + l15 * I_ + k1);
        ahn = ldsA(hA + l15 * H_ + k1);
      }
      acc = __builtin_amdgcn_wmma_f32_16x16x32_bf16(
          false, ax, false, *(const v16bf*)(wir + kk * 32 + lhi * 16),
          (short)0, acc, false, false);
      acc = __builtin_amdgcn_wmma_f32_16x16x32_bf16(
          false, ah, false, whh[kk], (short)0, acc, false, false);
      ax = axn; ah = ahn;
    }

    // C layout: VGPR r, lanes0-15 -> (M=r, N=lane); lanes16-31 -> (M=8+r).
#pragma unroll
    for (int r = 0; r < 8; ++r)
      gbuf[gate][r + lhi * 8][l15] = acc[r] + bsum;
    __syncthreads();

    // ---- pointwise LSTM cell update (block-local) ----
    for (int e = tid; e < B_ * 16; e += NTHR) {
      const int m = e >> 4, c = e & 15;
      float ig = sigm(gbuf[0][m][c]);
      float fg = sigm(gbuf[1][m][c]);
      float gg = tanhf(gbuf[2][m][c]);
      float og = sigm(gbuf[3][m][c]);
      float cn = fg * cst[m][c] + ig * gg;
      float hn = og * tanhf(cn);
      cst[m][c] = cn;
      hl[m][c]  = hn;
      out[((size_t)m * T_ + t) * H_ + blk * 16 + c] = hn;
      h_glb[m * H_ + blk * 16 + c] = (__bf16)hn;
    }
    __threadfence();        // publish this block's h slice (device scope)
    __syncthreads();

    // ---- grid-wide split barrier via monotonic counters ----
    if (tid == 0) {
      unsigned a = __hip_atomic_fetch_add(arrive, 1u, __ATOMIC_ACQ_REL,
                                          __HIP_MEMORY_SCOPE_AGENT);
      if (a == (unsigned)((t + 1) * NBLK - 1))
        __hip_atomic_store(release, (unsigned)(t + 1), __ATOMIC_RELEASE,
                           __HIP_MEMORY_SCOPE_AGENT);
      while (__hip_atomic_load(release, __ATOMIC_ACQUIRE,
                               __HIP_MEMORY_SCOPE_AGENT) < (unsigned)(t + 1))
        __builtin_amdgcn_s_sleep(2);
    }
    __syncthreads();
    __threadfence();        // acquire: invalidate stale h_glb lines in WGP$

    // ---- async reload of full h into LDS; drained at next loop top ----
    for (int i = tid * 8; i < B_ * H_; i += NTHR * 8)
      async_cp16(hoff + (unsigned)i * 2u, (unsigned)(i * 2), h_glb);
  }

  wait_async0();            // retire dangling h reload from the last step
  __syncthreads();
  // ---- h_n [1,B,H] and c_n [1,B,H] appended after output [B,T,H] ----
  const size_t o1 = (size_t)B_ * T_ * H_;
  for (int e = tid; e < B_ * 16; e += NTHR) {
    const int m = e >> 4, c = e & 15;
    out[o1 + (size_t)m * H_ + blk * 16 + c]                   = hl[m][c];
    out[o1 + (size_t)B_ * H_ + (size_t)m * H_ + blk * 16 + c] = cst[m][c];
  }
}

extern "C" void kernel_launch(void* const* d_in, const int* in_sizes, int n_in,
                              void* d_out, int out_size, void* d_ws, size_t ws_size,
                              hipStream_t stream) {
  const float* x    = (const float*)d_in[0];   // [B,T,I]
  const float* w_ih = (const float*)d_in[1];   // [4H,I]
  const float* w_hh = (const float*)d_in[2];   // [4H,H]
  const float* b_ih = (const float*)d_in[3];   // [4H]
  const float* b_hh = (const float*)d_in[4];   // [4H]
  float* out = (float*)d_out;

  // Workspace layout:
  //   [0,16K)      h_glb (bf16 16x512)
  //   [16K,16K+8)  arrive/release counters
  //   [32K,32K+2M) W_ih bf16
  //   [4M,4M+64M)  x bf16
  __bf16*   h_glb  = (__bf16*)d_ws;
  unsigned* ctr    = (unsigned*)((char*)d_ws + 16384);
  __bf16*   wih_bf = (__bf16*)((char*)d_ws + 32768);
  __bf16*   x_bf   = (__bf16*)((char*)d_ws + (size_t)4 * 1024 * 1024);

  init_ctr_kernel<<<1, 1, 0, stream>>>(ctr);
  const int nwih = 4 * H_ * I_;                    // 1,048,576
  cvt_bf16_kernel<<<nwih / 1024, 256, 0, stream>>>(w_ih, wih_bf, nwih);
  const int nx = B_ * T_ * I_;                     // 33,554,432
  cvt_bf16_kernel<<<nx / 1024, 256, 0, stream>>>(x, x_bf, nx);
  lstm_scan_kernel<<<NBLK, NTHR, 0, stream>>>(x_bf, w_hh, b_ih, b_hh, wih_bf,
                                              out, h_glb, ctr);
}
